// QuantLinearW4A4_57208964383082
// MI455X (gfx1250) — 8-bit matrix operand order repaired and checked by reading, compiled, not run on hardware
//
#include <hip/hip_runtime.h>


typedef int   v8i __attribute__((ext_vector_type(8)));
typedef int   v4i __attribute__((ext_vector_type(4)));
typedef float v8f __attribute__((ext_vector_type(8)));
typedef float v2f __attribute__((ext_vector_type(2)));
typedef unsigned long long u64;

typedef int b64v __attribute__((vector_size(8)));
typedef __attribute__((address_space(1))) b64v* glb64_t;
typedef __attribute__((address_space(3))) b64v* lds64_t;

constexpr int T_   = 8192;
constexpr int IN_  = 4096;
constexpr int OUT_ = 4096;
constexpr int R_   = 64;

constexpr int BM = 128;
constexpr int BN = 256;
constexpr int BK = 64;
constexpr int LDST = 80;
constexpr int ASTG = BM * LDST;
constexpr int BSTG = BN * LDST;

#if __has_builtin(__builtin_amdgcn_global_load_async_to_lds_b64)
#define USE_ASYNC 1
#else
#define USE_ASYNC 0
#endif

__device__ __forceinline__ void copy_b64_g2l(const unsigned char* g,
                                             unsigned char* l) {
#if USE_ASYNC
  __builtin_amdgcn_global_load_async_to_lds_b64(
      (glb64_t)g, (lds64_t)l,  0,  0);
#else
  *(u64*)l = *(const u64*)g;
#endif
}

__device__ __forceinline__ void wait_async_all() {
#if USE_ASYNC
#if __has_builtin(__builtin_amdgcn_s_wait_asynccnt)
  __builtin_amdgcn_s_wait_asynccnt(0);
#else
  asm volatile("s_wait_asynccnt 0x0" ::: "memory");
#endif
#endif
}

__global__ __launch_bounds__(256) void pack_i32_to_i8(
    const int* __restrict__ src, signed char* __restrict__ dst, int n4) {
  int i = blockIdx.x * blockDim.x + threadIdx.x;
  if (i < n4) {
    int4 v = ((const int4*)src)[i];
    char4 c = make_char4((signed char)v.x, (signed char)v.y,
                         (signed char)v.z, (signed char)v.w);
    ((char4*)dst)[i] = c;
  }
}

__global__ __launch_bounds__(256) void w4a4_wmma_gemm(
    const unsigned char* __restrict__ pkX,
    const unsigned char* __restrict__ pkW,
    const float* __restrict__ scale_x,
    const float* __restrict__ weight_scales,
    const float* __restrict__ outlier_x,
    const float* __restrict__ w_outlier,
    const float* __restrict__ bias,
    float* __restrict__ out)
{
  __shared__ alignas(16) unsigned char sA[2 * ASTG];
  __shared__ alignas(16) unsigned char sB[2 * BSTG];

  const int t   = threadIdx.x;
  const int l   = t & 31;
  const int m16 = l & 15;
  const int h   = l >> 4;
  const int wid = t >> 5;
  const int wM  = wid >> 2;
  const int wN  = wid & 3;
  const int bm0 = blockIdx.y * BM;
  const int bn0 = blockIdx.x * BN;

  v8i acc[4][4];
#pragma unroll
  for (int mf = 0; mf < 4; ++mf)
#pragma unroll
    for (int nf = 0; nf < 4; ++nf)
      acc[mf][nf] = (v8i){0, 0, 0, 0, 0, 0, 0, 0};

  auto stage = [&](int buf, int k0) {
#pragma unroll
    for (int i = 0; i < 4; ++i) {
      int c   = t + i * 256;
      int row = c >> 3;
      int col = (c & 7) << 3;
      copy_b64_g2l(pkX + (size_t)(bm0 + row) * IN_ + k0 + col,
                   sA + buf * ASTG + row * LDST + col);
    }
#pragma unroll
    for (int i = 0; i < 8; ++i) {
      int c   = t + i * 256;
      int row = c >> 3;
      int col = (c & 7) << 3;
      copy_b64_g2l(pkW + (size_t)(bn0 + row) * IN_ + k0 + col,
                   sB + buf * BSTG + row * LDST + col);
    }
  };

  stage(0, 0);
  int buf = 0;

  for (int k0 = 0; k0 < IN_; k0 += BK) {
    wait_async_all();
    __syncthreads();
    if (k0 + BK < IN_) stage(buf ^ 1, k0 + BK);

    const unsigned char* bufA = sA + buf * ASTG;
    const unsigned char* bufB = sB + buf * BSTG;

    v8i aF[4];
#pragma unroll
    for (int mf = 0; mf < 4; ++mf) {
      const unsigned char* p = bufA + (wM * 64 + mf * 16 + m16) * LDST + 8 * h;
      union { u64 q[4]; v8i v; } u;
      u.q[0] = *(const u64*)(p);
      u.q[1] = *(const u64*)(p + 16);
      u.q[2] = *(const u64*)(p + 32);
      u.q[3] = *(const u64*)(p + 48);
      aF[mf] = u.v;
    }
#pragma unroll
    for (int nf = 0; nf < 4; ++nf) {
      const unsigned char* p = bufB + (wN * 64 + nf * 16 + m16) * LDST + 8 * h;
      union { u64 x[4]; v8i v; } u;
      u.x[0] = *(const u64*)(p);
      u.x[1] = *(const u64*)(p + 16);
      u.x[2] = *(const u64*)(p + 32);
      u.x[3] = *(const u64*)(p + 48);
      v8i bF = u.v;
#pragma unroll
      for (int mf = 0; mf < 4; ++mf)
        acc[mf][nf] = __builtin_amdgcn_wmma_i32_16x16x64_iu8(
             true, aF[mf],  true, bF,
            acc[mf][nf],  false,  false);
    }
    buf ^= 1;
  }

#pragma unroll
  for (int mf = 0; mf < 4; ++mf) {
    const int gm0 = bm0 + wM * 64 + mf * 16;
    float sx[8];
#pragma unroll
    for (int j = 0; j < 8; ++j) sx[j] = scale_x[gm0 + j + 8 * h];

    v2f av[16];
    const float* oxrow = outlier_x + (size_t)(gm0 + m16) * R_;
#pragma unroll
    for (int r = 0; r < 16; ++r)
      av[r] = *(const v2f*)(oxrow + 4 * r + 2 * h);

#pragma unroll
    for (int nf = 0; nf < 4; ++nf) {
      const int cn = bn0 + wN * 64 + nf * 16 + m16;
      const float* worow = w_outlier + (size_t)cn * R_;
      const float bcol = bias[cn];
      v8f cf;
#pragma unroll
      for (int j = 0; j < 8; ++j) cf[j] = bcol;
#pragma unroll
      for (int r = 0; r < 16; ++r) {
        v2f bv = *(const v2f*)(worow + 4 * r + 2 * h);
        cf = __builtin_amdgcn_wmma_f32_16x16x4_f32(
            false, av[r], false, bv, (short)0, cf, false, false);
      }
      const float swc = weight_scales[cn];
#pragma unroll
      for (int j = 0; j < 8; ++j) {
        float v = (float)acc[mf][nf][j] * sx[j] * swc + cf[j];
        out[(size_t)(gm0 + j + 8 * h) * OUT_ + cn] = v;
      }
    }
  }
}

extern "C" void kernel_launch(void* const* d_in, const int* in_sizes, int n_in,
                              void* d_out, int out_size, void* d_ws, size_t ws_size,
                              hipStream_t stream) {
  const int*   quant_x       = (const int*)d_in[0];
  const int*   weight_q      = (const int*)d_in[1];
  const float* scale_x       = (const float*)d_in[2];
  const float* weight_scales = (const float*)d_in[3];
  const float* outlier_x     = (const float*)d_in[4];
  const float* w_outlier     = (const float*)d_in[5];
  const float* bias          = (const float*)d_in[6];
  float*       out           = (float*)d_out;

  signed char* pkX = (signed char*)d_ws;
  signed char* pkW = pkX + (size_t)T_ * IN_;

  const int nx4 = T_ * IN_ / 4;
  const int nw4 = OUT_ * IN_ / 4;
  pack_i32_to_i8<<<nx4 / 256, 256, 0, stream>>>(quant_x, pkX, nx4);
  pack_i32_to_i8<<<nw4 / 256, 256, 0, stream>>>(weight_q, pkW, nw4);

  dim3 grid(OUT_ / BN, T_ / BM);
  w4a4_wmma_gemm<<<grid, 256, 0, stream>>>(
      (const unsigned char*)pkX, (const unsigned char*)pkW,
      scale_x, weight_scales, outlier_x, w_outlier, bias, out);
}
